// DynamicEdgeConv_56281251446899
// MI455X (gfx1250) — compile-verified, matrix operand order checked by reading, not run on hardware
//
#include <hip/hip_runtime.h>
#include <cstdint>
#include <cstddef>

// =====================================================================
// DynamicEdgeConv for MI455X (gfx1250, wave32).
// bf16 WMMA (v_wmma_f32_16x16x32_bf16) GEMMs with fused bias+BN+ReLU and
// fused max-over-K neighbor pooling in the second GEMM of each layer.
// Workspace requirement: ~115 MB.
// =====================================================================

#define NPTS 12288
#define KNN  16
#define EPSV 1e-5f

typedef __attribute__((ext_vector_type(16))) __bf16 v16bf;
typedef __attribute__((ext_vector_type(8)))  float  v8f;

// ---------------------------------------------------------------------
// Load a 16x32 bf16 fragment (A layout; B uses W^T so the pattern is the
// same) from an LDS tile stored row-major with row stride `stride`.
// Lane pattern per CDNA5 ISA 7.12.2:
//   lanes 0-15  : row = lane,    K in {0..7, 16..23}
//   lanes 16-31 : row = lane-16, K in {8..15, 24..31}
// ---------------------------------------------------------------------
__device__ __forceinline__ v16bf load_frag(const __bf16* __restrict__ base, int stride) {
  const int lane = threadIdx.x & 31;
  const int r    = lane & 15;
  const int kh   = (lane >> 4) << 3;          // 0 or 8
  v16bf f;
#pragma unroll
  for (int i = 0; i < 8; ++i) {
    const int k = ((i >> 2) << 4) + kh + ((i & 3) << 1);
    f[2 * i]     = base[r * stride + k];
    f[2 * i + 1] = base[r * stride + k + 1];
  }
  return f;
}

// ---------------------------------------------------------------------
// Unified GEMM + bias + BN + ReLU kernel.
//   MODE 0 (EDGE) : A rows built on the fly as [xi, xj - xi] from the
//                   previous layer activations + knn indices. Output is
//                   the full (N*K, HOUT) bf16 activation buffer h1.
//   MODE 1 (PLAIN): A rows read from h1; epilogue takes max over the 16
//                   neighbor rows of each point -> (N, HOUT) bf16.
//   MODE 2 (OUT)  : A rows are per-point activations (M = N); epilogue
//                   max-reduces the 16-row tile and atomicMax's into the
//                   global pooled[32] vector.
// KDIM = padded K dimension (multiple of 32), HOUT = output columns.
// WPB  = waves per block; each wave owns one 16-row M tile.
// ---------------------------------------------------------------------
template <int KDIM, int HOUT, int MODE, int WPB>
__global__ __launch_bounds__(WPB * 32, 1)
void gemm_bn_kernel(const __bf16* __restrict__ Ain,
                    const int*    __restrict__ nbr,
                    const __bf16* __restrict__ Wt,     // bf16 W^T: [HOUT][KDIM]
                    const float*  __restrict__ bias,
                    const float*  __restrict__ gain,
                    const float*  __restrict__ beta,
                    const float*  __restrict__ mean,
                    const float*  __restrict__ var,
                    __bf16*       __restrict__ obf,
                    float*        __restrict__ pooled,
                    int cin) {
  extern __shared__ __bf16 smem[];
  __bf16* sB = smem;                       // [HOUT][KDIM]
  __bf16* sA = smem + HOUT * KDIM;         // [WPB][16][KDIM]

  const int tid  = threadIdx.x;
  const int lane = tid & 31;
  const int w    = tid >> 5;
  const int tile = blockIdx.x * WPB + w;   // EDGE/PLAIN: point id; OUT: 16-point tile

  // Cooperative stage of W^T into LDS (dword copies).
  {
    const uint32_t* src = (const uint32_t*)Wt;
    uint32_t*       dst = (uint32_t*)sB;
    for (int i = tid; i < HOUT * KDIM / 2; i += WPB * 32) dst[i] = src[i];
  }

  // Fill this wave's private A tile.
  __bf16* myA = sA + w * 16 * KDIM;
  if (MODE == 0) {
    for (int e = lane; e < 16 * KDIM; e += 32) {
      const int r = e / KDIM;              // neighbor slot (KDIM is pow2)
      const int c = e & (KDIM - 1);
      float v = 0.f;
      if (c < cin) {
        v = (float)Ain[tile * cin + c];
      } else if (c < 2 * cin) {
        const int j  = nbr[tile * KNN + r];
        const int cc = c - cin;
        v = (float)Ain[j * cin + cc] - (float)Ain[tile * cin + cc];
      }
      myA[e] = (__bf16)v;
    }
  } else {
    // 16 contiguous rows of A -> straight dword copy into LDS.
    const uint32_t* src = (const uint32_t*)(Ain + (size_t)tile * 16 * KDIM);
    uint32_t*       dst = (uint32_t*)myA;
    __builtin_prefetch(Ain + ((size_t)tile + WPB) * 16 * KDIM, 0, 1);
    for (int i = lane; i < 16 * KDIM / 2; i += 32) dst[i] = src[i];
  }
  __syncthreads();

  const int coll = lane & 15;

#pragma unroll 1
  for (int nt = 0; nt < HOUT / 16; ++nt) {
    v8f acc = {0.f, 0.f, 0.f, 0.f, 0.f, 0.f, 0.f, 0.f};
#pragma unroll
    for (int k0 = 0; k0 < KDIM / 32; ++k0) {
      v16bf a = load_frag(myA + k0 * 32, KDIM);
      v16bf b = load_frag(sB + nt * 16 * KDIM + k0 * 32, KDIM);
      acc = __builtin_amdgcn_wmma_f32_16x16x32_bf16(
          /*neg_a=*/false, a, /*neg_b=*/false, b,
          /*c_mod=*/(short)0, acc, /*reuse_a=*/false, /*reuse_b=*/false);
    }

    // Fused bias + BN + ReLU:  y = relu(z*s + t)
    const int   c = nt * 16 + coll;
    const float s = gain[c] * rsqrtf(var[c] + EPSV);
    const float t = (bias[c] - mean[c]) * s + beta[c];
    float vals[8];
#pragma unroll
    for (int i = 0; i < 8; ++i) {
      const float v = fmaf(acc[i], s, t);
      vals[i] = v > 0.f ? v : 0.f;
    }

    if (MODE == 0) {
      // Store full edge activations (row = tile*16 + m).
      const int mbase = (lane >> 4) << 3;
#pragma unroll
      for (int i = 0; i < 8; ++i)
        obf[(size_t)(tile * 16 + mbase + i) * HOUT + c] = (__bf16)vals[i];
    } else {
      // Max over the 16 rows of this tile (M dimension of the C tile).
      float m = vals[0];
#pragma unroll
      for (int i = 1; i < 8; ++i) m = m > vals[i] ? m : vals[i];
      const float o = __shfl_xor(m, 16, 32);   // combine the two M halves
      m = m > o ? m : o;
      if (MODE == 1) {
        if (lane < 16) obf[(size_t)tile * HOUT + c] = (__bf16)m;
      } else {
        // values >= 0 after ReLU -> uint compare == float compare
        if (lane < 16) atomicMax((unsigned int*)&pooled[c], __float_as_uint(m));
      }
    }
  }
}

// ---------------------------------------------------------------------
// KNN: one thread per query row, pos tiles staged in LDS, fully
// register-resident sorted top-16 (bubble insert, static indices).
// ---------------------------------------------------------------------
__global__ __launch_bounds__(256, 1)
void knn_kernel(const float* __restrict__ pos, int* __restrict__ nbr) {
  __shared__ float sp[256][3];
  const int row = blockIdx.x * 256 + threadIdx.x;
  const float px = pos[row * 3 + 0], py = pos[row * 3 + 1], pz = pos[row * 3 + 2];
  float bd[KNN];
  int   bi[KNN];
#pragma unroll
  for (int i = 0; i < KNN; ++i) { bd[i] = 3.4e38f; bi[i] = 0; }

  for (int tb = 0; tb < NPTS; tb += 256) {
    const int j = tb + threadIdx.x;
    sp[threadIdx.x][0] = pos[j * 3 + 0];
    sp[threadIdx.x][1] = pos[j * 3 + 1];
    sp[threadIdx.x][2] = pos[j * 3 + 2];
    __syncthreads();
    for (int t = 0; t < 256; ++t) {
      const int   jj = tb + t;
      const float dx = px - sp[t][0];
      const float dy = py - sp[t][1];
      const float dz = pz - sp[t][2];
      float d = fmaf(dx, dx, fmaf(dy, dy, dz * dz));
      if (jj == row) d = 3.5e38f;
      if (d < bd[KNN - 1]) {
        bd[KNN - 1] = d; bi[KNN - 1] = jj;
#pragma unroll
        for (int q = KNN - 1; q > 0; --q) {
          if (bd[q] < bd[q - 1]) {
            const float td = bd[q]; bd[q] = bd[q - 1]; bd[q - 1] = td;
            const int   ti = bi[q]; bi[q] = bi[q - 1]; bi[q - 1] = ti;
          }
        }
      }
    }
    __syncthreads();
  }
#pragma unroll
  for (int i = 0; i < KNN; ++i) nbr[row * KNN + i] = bi[i];
}

// ---------------------------------------------------------------------
// Prep / epilogue helper kernels.
// ---------------------------------------------------------------------
__global__ void conv_wt_kernel(const float* __restrict__ W, __bf16* __restrict__ out,
                               int Ksrc, int Kdst, int Nout) {
  const int i = blockIdx.x * 256 + threadIdx.x;
  if (i >= Nout * Kdst) return;
  const int n = i / Kdst, k = i % Kdst;
  out[i] = (k < Ksrc) ? (__bf16)W[k * Nout + n] : (__bf16)0.f;  // W^T, K-padded
}

__global__ void conv_x_kernel(const float* __restrict__ x, __bf16* __restrict__ o, int n) {
  const int i = blockIdx.x * 256 + threadIdx.x;
  if (i < n) o[i] = (__bf16)x[i];
}

__global__ void zero32_kernel(float* p) { p[threadIdx.x] = 0.f; }

__global__ void final_fc_kernel(const float* __restrict__ pooled,
                                const float* __restrict__ Wf,
                                const float* __restrict__ bfv,
                                float* __restrict__ out) {
  const int c = threadIdx.x;  // 32 threads
  float s = bfv[c];
#pragma unroll
  for (int j = 0; j < 32; ++j) s = fmaf(pooled[j], Wf[j * 32 + c], s);
  out[c] = s;
}

// ---------------------------------------------------------------------
// Host launcher.
// Input mapping: setup_inputs() insertion order for x/pos/batch, then
// `params` flattened in JAX pytree (sorted-key) order:
//   3:Wf 4:Wo 5:beo 6:bf 7:bo 8:go
//   9+12L+j for layer L, j: 0:W1 1:W2 2:b1 3:b2 4:be1 5:be2 6:g1 7:g2
//                           8:m1 9:m2 10:v1 11:v2
//   45:mo 46:vo
// ---------------------------------------------------------------------
#define LP(L, j) ((const float*)d_in[9 + 12 * (L) + (j)])

extern "C" void kernel_launch(void* const* d_in, const int* in_sizes, int n_in,
                              void* d_out, int out_size, void* d_ws, size_t ws_size,
                              hipStream_t stream) {
  (void)in_sizes; (void)n_in; (void)out_size; (void)ws_size;

  const float* x   = (const float*)d_in[0];
  const float* pos = (const float*)d_in[1];
  const float* Wf  = (const float*)d_in[3];
  const float* Wo  = (const float*)d_in[4];
  const float* beo = (const float*)d_in[5];
  const float* bfv = (const float*)d_in[6];
  const float* bo  = (const float*)d_in[7];
  const float* go  = (const float*)d_in[8];
  const float* mo  = (const float*)d_in[45];
  const float* vo  = (const float*)d_in[46];

  // ---- workspace layout (bytes) ----
  char*   ws     = (char*)d_ws;
  int*    nbr    = (int*)ws;                                        //   786,432
  __bf16* xbA    = (__bf16*)(ws + 786432);                          // 6,291,456
  __bf16* xbB    = (__bf16*)(ws + 786432 + 6291456);                // 6,291,456
  __bf16* h1     = (__bf16*)(ws + 786432 + 2 * 6291456);            // 100,663,296
  __bf16* wb     = (__bf16*)(ws + 786432 + 2 * 6291456 + 100663296);//   356,352
  float*  pooled = (float*)(ws + 786432 + 2 * 6291456 + 100663296 + 356352);

  __bf16* l1w1 = wb;            // 64  x 32
  __bf16* l1w2 = wb + 2048;     // 64  x 64
  __bf16* l2w1 = wb + 6144;     // 128 x 128
  __bf16* l2w2 = wb + 22528;    // 128 x 128
  __bf16* l3w1 = wb + 38912;    // 256 x 256
  __bf16* l3w2 = wb + 104448;   // 256 x 256
  __bf16* wot  = wb + 169984;   // 32  x 256

  // ---- weight / input conversion to bf16 (transposed, K-padded) ----
  conv_wt_kernel<<<(64 * 32 + 255) / 256, 256, 0, stream>>>(LP(0, 0), l1w1, 6, 32, 64);
  conv_wt_kernel<<<(64 * 64 + 255) / 256, 256, 0, stream>>>(LP(0, 1), l1w2, 64, 64, 64);
  conv_wt_kernel<<<(128 * 128) / 256, 256, 0, stream>>>(LP(1, 0), l2w1, 128, 128, 128);
  conv_wt_kernel<<<(128 * 128) / 256, 256, 0, stream>>>(LP(1, 1), l2w2, 128, 128, 128);
  conv_wt_kernel<<<(256 * 256) / 256, 256, 0, stream>>>(LP(2, 0), l3w1, 256, 256, 256);
  conv_wt_kernel<<<(256 * 256) / 256, 256, 0, stream>>>(LP(2, 1), l3w2, 256, 256, 256);
  conv_wt_kernel<<<(32 * 256) / 256, 256, 0, stream>>>(Wo, wot, 256, 256, 32);
  conv_x_kernel<<<(NPTS * 3 + 255) / 256, 256, 0, stream>>>(x, xbA, NPTS * 3);

  // ---- KNN ----
  knn_kernel<<<NPTS / 256, 256, 0, stream>>>(pos, nbr);

  // ---- Layer 1 (c=3 -> 64) ----
  gemm_bn_kernel<32, 64, 0, 8><<<NPTS / 8, 256, (64 * 32 + 8 * 16 * 32) * 2, stream>>>(
      xbA, nbr, l1w1, LP(0, 2), LP(0, 6), LP(0, 4), LP(0, 8), LP(0, 10), h1, nullptr, 3);
  gemm_bn_kernel<64, 64, 1, 8><<<NPTS / 8, 256, (64 * 64 + 8 * 16 * 64) * 2, stream>>>(
      h1, nullptr, l1w2, LP(0, 3), LP(0, 7), LP(0, 5), LP(0, 9), LP(0, 11), xbB, nullptr, 0);

  // ---- Layer 2 (64 -> 128) ----
  gemm_bn_kernel<128, 128, 0, 8><<<NPTS / 8, 256, (128 * 128 + 8 * 16 * 128) * 2, stream>>>(
      xbB, nbr, l2w1, LP(1, 2), LP(1, 6), LP(1, 4), LP(1, 8), LP(1, 10), h1, nullptr, 64);
  gemm_bn_kernel<128, 128, 1, 8><<<NPTS / 8, 256, (128 * 128 + 8 * 16 * 128) * 2, stream>>>(
      h1, nullptr, l2w2, LP(1, 3), LP(1, 7), LP(1, 5), LP(1, 9), LP(1, 11), xbA, nullptr, 0);

  // ---- Layer 3 (128 -> 256); WPB=4 keeps LDS at 160 KB ----
  gemm_bn_kernel<256, 256, 0, 4><<<NPTS / 4, 128, (256 * 256 + 4 * 16 * 256) * 2, stream>>>(
      xbA, nbr, l3w1, LP(2, 2), LP(2, 6), LP(2, 4), LP(2, 8), LP(2, 10), h1, nullptr, 128);
  gemm_bn_kernel<256, 256, 1, 4><<<NPTS / 4, 128, (256 * 256 + 4 * 16 * 256) * 2, stream>>>(
      h1, nullptr, l3w2, LP(2, 3), LP(2, 7), LP(2, 5), LP(2, 9), LP(2, 11), xbB, nullptr, 0);

  // ---- Output head + global max pool + final FC ----
  zero32_kernel<<<1, 32, 0, stream>>>(pooled);
  gemm_bn_kernel<256, 32, 2, 8><<<NPTS / 16 / 8, 256, (32 * 256 + 8 * 16 * 256) * 2, stream>>>(
      xbB, nullptr, wot, bo, go, beo, mo, vo, nullptr, pooled, 0);
  final_fc_kernel<<<1, 32, 0, stream>>>(pooled, Wf, bfv, (float*)d_out);
}
